// AccuracyEmbeddingWrapper_42133629174011
// MI455X (gfx1250) — matrix operand order repaired and checked by reading, not compiled, not run on hardware
//
#include <hip/hip_runtime.h>
#include <hip/hip_bf16.h>

// ---------------------------------------------------------------------------
// kNN-accuracy: count-beats-target formulation.
//   hit[n] = ( #{ v : s[n,v] > s[n,target[n]] } <= K-1 )
// s[n,v] = x_n . w_v - 0.5*||w_v||^2   (rank-equivalent to -d2)
// GEMM on v_wmma_f32_16x16x32_bf16, split-bf16 (hi+lo) 3-term fp32 emulation.
// X block staged global->LDS via async loads (ASYNCcnt); W chunks double-
// buffered in LDS with register prefetch overlapping the WMMA loop.
// ---------------------------------------------------------------------------

#define N_ROWS 4096
#define DQ     512
#define VW     50000
#define KTOP   10

#define MB   64          // rows per block
#define VCH  32          // v columns per chunk
#define XS   520         // padded LDS stride (halves) for X block
#define WSTR 520         // padded LDS stride (halves) for W chunk
#define CHUNKS_PER_BLOCK 16

typedef __attribute__((ext_vector_type(16))) __bf16 v16bf;
typedef __attribute__((ext_vector_type(8)))  float  v8f;

union FragB { uint4 q[2]; v16bf v; };

__device__ __forceinline__ unsigned hi16(float f) {
    return __float_as_uint(f) >> 16;           // truncate to bf16 (residual -> lo)
}
__device__ __forceinline__ unsigned lo16(float f) {
    float hif = __uint_as_float(__float_as_uint(f) & 0xFFFF0000u);
    return __float_as_uint(f - hif) >> 16;
}

// CDNA5 async global->LDS copy, 16 bytes per lane, tracked by ASYNCcnt.
__device__ __forceinline__ void async_ld128(unsigned lds_off, const void* gaddr) {
    asm volatile("global_load_async_to_lds_b128 %0, %1, off"
                 :: "v"(lds_off), "v"(gaddr) : "memory");
}
__device__ __forceinline__ void wait_async0() {
    asm volatile("s_wait_asynccnt 0x0" ::: "memory");
}

// ---------------------------------------------------------------------------
// Kernel 1: split logits into bf16 hi/lo planes; zero per-row counters.
// ---------------------------------------------------------------------------
__global__ __launch_bounds__(256) void knn_prep_kernel(
    const float* __restrict__ x, unsigned short* __restrict__ xhi,
    unsigned short* __restrict__ xlo, unsigned int* __restrict__ cnt)
{
    int i = blockIdx.x * 256 + threadIdx.x;
    if (i < N_ROWS * DQ) {
        float f = x[i];
        xhi[i] = (unsigned short)hi16(f);
        xlo[i] = (unsigned short)lo16(f);
    }
    if (i < N_ROWS) cnt[i] = 0u;
}

// ---------------------------------------------------------------------------
// Kernel 2: target scores t[n] = x_n . w_t - 0.5*||w_t||^2  (f32, one wave/row)
// ---------------------------------------------------------------------------
__global__ __launch_bounds__(256) void knn_tscore_kernel(
    const float* __restrict__ x, const int* __restrict__ target,
    const float* __restrict__ wv, float* __restrict__ t)
{
    int wave = threadIdx.x >> 5, lane = threadIdx.x & 31;
    int n = blockIdx.x * 8 + wave;
    if (n >= N_ROWS) return;
    const float* xr = x + (size_t)n * DQ;
    const float* wr = wv + (size_t)target[n] * DQ;
    float dot = 0.f, w2 = 0.f;
    for (int k = lane; k < DQ; k += 32) {
        float a = xr[k], b = wr[k];
        dot += a * b; w2 += b * b;
    }
#pragma unroll
    for (int off = 16; off; off >>= 1) {
        dot += __shfl_xor(dot, off, 32);
        w2  += __shfl_xor(w2,  off, 32);
    }
    if (lane == 0) t[n] = dot - 0.5f * w2;
}

// ---------------------------------------------------------------------------
// Kernel 3: fused WMMA GEMM + beats-target counting.
// Grid: x = N/64 row blocks, y = chunk groups (16 chunks of 32 cols each).
// ---------------------------------------------------------------------------
__global__ __launch_bounds__(256) void knn_count_kernel(
    const unsigned short* __restrict__ xhi, const unsigned short* __restrict__ xlo,
    const float* __restrict__ wv, const float* __restrict__ tsc,
    const int* __restrict__ target, unsigned int* __restrict__ gcount)
{
    __shared__ __align__(16) unsigned short xs_hi[MB * XS];
    __shared__ __align__(16) unsigned short xs_lo[MB * XS];
    __shared__ __align__(16) unsigned short ws_hi[2][VCH * WSTR];
    __shared__ __align__(16) unsigned short ws_lo[2][VCH * WSTR];
    __shared__ float        w2_s[2][VCH];    // 0.5*||w||^2 per chunk column
    __shared__ float        t_s[MB];
    __shared__ int          tg_s[MB];
    __shared__ unsigned int cnt_s[MB];

    const int tid = threadIdx.x;
    const int m0  = blockIdx.x * MB;

    // --- async-stage X block (bf16 hi/lo) into LDS: 4 threads/row, 128 halves
    {
        int row = tid >> 2;
        int c0  = (tid & 3) * 128;
        const unsigned short* sh = xhi + (size_t)(m0 + row) * DQ + c0;
        const unsigned short* sl = xlo + (size_t)(m0 + row) * DQ + c0;
        unsigned dh = (unsigned)(size_t)(xs_hi + row * XS + c0);
        unsigned dl = (unsigned)(size_t)(xs_lo + row * XS + c0);
#pragma unroll
        for (int j = 0; j < 16; ++j) {            // 16B per async op
            async_ld128(dh + 16u * j, sh + 8 * j);
            async_ld128(dl + 16u * j, sl + 8 * j);
        }
    }
    if (tid < MB) {
        t_s[tid]   = tsc[m0 + tid];
        tg_s[tid]  = target[m0 + tid];
        cnt_s[tid] = 0u;
    }

    // --- prefetch W chunk 0 into registers (8 threads/row, 64 floats each)
    const int wrow = tid >> 3;
    const int wc0  = (tid & 7) * 64;
    float4 wreg[16];
    {
        int vr = blockIdx.y * CHUNKS_PER_BLOCK * VCH + wrow;
        if (vr > VW - 1) vr = VW - 1;
        const float4* src = (const float4*)(wv + (size_t)vr * DQ + wc0);
#pragma unroll
        for (int j = 0; j < 16; ++j) wreg[j] = src[j];
    }

    wait_async0();          // X block landed in LDS
    __syncthreads();

    const int wave = tid >> 5;
    const int lane = tid & 31;
    const int half = lane >> 4;
    const int ln   = lane & 15;
    const int wm   = wave & 3;       // M subtile 0..3
    const int wvt  = wave >> 2;      // V subtile 0..1
    const int arow = wm * 16 + ln;
    const int brow = wvt * 16 + ln;
    const int koffA = half * 8;      // A frag: K pairs at koffA and koffA+16
    const int koffB = half * 8;      // B frag: K pairs at koffB and koffB+16

    float tv[8]; int tg[8]; int cnt[8];
#pragma unroll
    for (int r = 0; r < 8; ++r) {
        int rl = wm * 16 + r + 8 * half;
        tv[r] = t_s[rl]; tg[r] = tg_s[rl]; cnt[r] = 0;
    }

    for (int ch = 0; ch < CHUNKS_PER_BLOCK; ++ch) {
        int vbase = (blockIdx.y * CHUNKS_PER_BLOCK + ch) * VCH;
        if (vbase >= VW) break;
        const int buf = ch & 1;

        // --- convert prefetched regs -> bf16 hi/lo LDS buffer + 0.5*||w||^2
        {
            float psum = 0.f;
#pragma unroll
            for (int j = 0; j < 16; ++j) {
                float4 f = wreg[j];
                psum += f.x * f.x + f.y * f.y + f.z * f.z + f.w * f.w;
                uint2 ph = { hi16(f.x) | (hi16(f.y) << 16), hi16(f.z) | (hi16(f.w) << 16) };
                uint2 pl = { lo16(f.x) | (lo16(f.y) << 16), lo16(f.z) | (lo16(f.w) << 16) };
                *(uint2*)&ws_hi[buf][wrow * WSTR + wc0 + 4 * j] = ph;
                *(uint2*)&ws_lo[buf][wrow * WSTR + wc0 + 4 * j] = pl;
            }
#pragma unroll
            for (int off = 4; off; off >>= 1) psum += __shfl_xor(psum, off, 32);
            if ((tid & 7) == 0) w2_s[buf][wrow] = 0.5f * psum;
        }
        __syncthreads();     // single barrier/iter: safe with double buffering

        // --- prefetch next chunk into registers while WMMAs run
        int vnext = vbase + VCH;
        if (ch + 1 < CHUNKS_PER_BLOCK && vnext < VW) {
            int vr = vnext + wrow; if (vr > VW - 1) vr = VW - 1;
            const float4* src = (const float4*)(wv + (size_t)vr * DQ + wc0);
#pragma unroll
            for (int j = 0; j < 16; ++j) wreg[j] = src[j];
        }

        // --- 16x16 tile GEMM over K=512, split-bf16 3-term
        v8f acc = {0.f, 0.f, 0.f, 0.f, 0.f, 0.f, 0.f, 0.f};
#pragma unroll 4
        for (int kk = 0; kk < DQ; kk += 32) {
            FragB ah, al, bh, bl;
            const uint4* pah = (const uint4*)(xs_hi + arow * XS + kk + koffA);
            const uint4* pal = (const uint4*)(xs_lo + arow * XS + kk + koffA);
            const uint4* pbh = (const uint4*)(&ws_hi[buf][brow * WSTR + kk + koffB]);
            const uint4* pbl = (const uint4*)(&ws_lo[buf][brow * WSTR + kk + koffB]);
            FragB t0, t1, t2, t3;
            t0.q[0] = pah[0]; t0.q[1] = pah[2];   // A: +16 halves
            t1.q[0] = pal[0]; t1.q[1] = pal[2];
            t2.q[0] = pbh[0]; t2.q[1] = pbh[2];   // B: +16 halves
            t3.q[0] = pbl[0]; t3.q[1] = pbl[2];
            ah = t0; al = t1; bh = t2; bl = t3;
            acc = __builtin_amdgcn_wmma_f32_16x16x32_bf16(false, ah.v, false, bh.v,
                                                          (short)0, acc, false, false);
            acc = __builtin_amdgcn_wmma_f32_16x16x32_bf16(false, ah.v, false, bl.v,
                                                          (short)0, acc, false, false);
            acc = __builtin_amdgcn_wmma_f32_16x16x32_bf16(false, al.v, false, bh.v,
                                                          (short)0, acc, false, false);
        }

        // --- compare vs target threshold, count strict beats
        int vcol = vbase + wvt * 16 + ln;
        bool vok = (vcol < VW);
        float w2c = w2_s[buf][wvt * 16 + ln];
#pragma unroll
        for (int r = 0; r < 8; ++r) {
            float s = acc[r] - w2c;
            if (vok && (s > tv[r]) && (vcol != tg[r])) cnt[r]++;
        }
    }

    // --- reduce per-row counts: lanes -> LDS -> global
#pragma unroll
    for (int r = 0; r < 8; ++r) {
        int rl = wm * 16 + r + 8 * half;
        atomicAdd(&cnt_s[rl], (unsigned int)cnt[r]);
    }
    __syncthreads();
    if (tid < MB) atomicAdd(&gcount[m0 + tid], cnt_s[tid]);
}

// ---------------------------------------------------------------------------
// Kernel 4: accuracy = sum(mask * (count < K)) / sum(mask)
// ---------------------------------------------------------------------------
__global__ __launch_bounds__(256) void knn_final_kernel(
    const unsigned int* __restrict__ cnt, const int* __restrict__ mask,
    float* __restrict__ out)
{
    __shared__ float sh[256], sm[256];
    float h = 0.f, m = 0.f;
    for (int n = threadIdx.x; n < N_ROWS; n += 256) {
        float mm = (float)mask[n];
        m += mm;
        if (cnt[n] < (unsigned)KTOP) h += mm;
    }
    sh[threadIdx.x] = h; sm[threadIdx.x] = m;
    __syncthreads();
    for (int s = 128; s; s >>= 1) {
        if (threadIdx.x < s) { sh[threadIdx.x] += sh[threadIdx.x + s];
                               sm[threadIdx.x] += sm[threadIdx.x + s]; }
        __syncthreads();
    }
    if (threadIdx.x == 0) out[0] = sh[0] / sm[0];
}

// ---------------------------------------------------------------------------
extern "C" void kernel_launch(void* const* d_in, const int* in_sizes, int n_in,
                              void* d_out, int out_size, void* d_ws, size_t ws_size,
                              hipStream_t stream)
{
    const float* logits = (const float*)d_in[0];
    const int*   target = (const int*)d_in[1];
    const int*   mask   = (const int*)d_in[2];
    const float* wv     = (const float*)d_in[3];
    float*       out    = (float*)d_out;

    // workspace layout
    unsigned short* xhi = (unsigned short*)d_ws;                 // N*D halves
    unsigned short* xlo = xhi + (size_t)N_ROWS * DQ;             // N*D halves
    float*          tsc = (float*)(xlo + (size_t)N_ROWS * DQ);   // N f32
    unsigned int*   cnt = (unsigned int*)(tsc + N_ROWS);         // N u32

    knn_prep_kernel<<<(N_ROWS * DQ + 255) / 256, 256, 0, stream>>>(logits, xhi, xlo, cnt);
    knn_tscore_kernel<<<N_ROWS / 8, 256, 0, stream>>>(logits, target, wv, tsc);

    int nchunks = (VW + VCH - 1) / VCH;                           // 1563
    dim3 grid(N_ROWS / MB, (nchunks + CHUNKS_PER_BLOCK - 1) / CHUNKS_PER_BLOCK); // 64 x 98
    knn_count_kernel<<<grid, 256, 0, stream>>>(xhi, xlo, wv, tsc, target, cnt);

    knn_final_kernel<<<1, 256, 0, stream>>>(cnt, mask, out);
}
